// FEDformer_53575422050362
// MI455X (gfx1250) — matrix operand order repaired and checked by reading, not compiled, not run on hardware
//
#include <hip/hip_runtime.h>

// ---------------- model constants ----------------
constexpr int Bc   = 32;
constexpr int Lc   = 1024;
constexpr int PRED = 256;
constexpr int CIN  = 32;
constexpr int DMc  = 512;
constexpr int DFFc = 2048;
constexpr int Hc   = 8;
constexpr int Ec   = 64;   // DM/H
constexpr int Mc   = 64;   // kept Fourier modes
constexpr int ELc  = 2;
constexpr int KMA  = 25;
constexpr int BL   = Bc * Lc;     // 32768

typedef __attribute__((ext_vector_type(16))) _Float16 v16h;
typedef __attribute__((ext_vector_type(8)))  _Float16 v8h;
typedef __attribute__((ext_vector_type(8)))  float    v8f;

// ================= generic strided batched WMMA GEMM =================
// C[g,m,n] = scale * sum_k A[g,m,k]*B[g,n,k]  (+bias[n]) (+res[g,m,n]) (relu?)
// REQUIREMENT: Kd % 32 == 0 (all call sites satisfy this).
struct GemmParams {
    const float* A; const float* Bw; float* C;
    const float* bias; const float* res;
    long a_b, a_m, a_k;
    long b_b, b_n, b_k;
    long c_b, c_m, c_n;
    long r_b, r_m, r_n;
    int  Mr, N, Kd, relu;
    float scale;
};

#define TBM 128
#define TBN 128
#define TBK 32
#define TBKP 40   // padded LDS row stride in halves (80B) -> conflict-free b128 reads
#define TILEA (TBM * TBKP)
#define TILEB (TBN * TBKP)

__device__ inline void cvt4(v16h& t, int o, float4 x) {
    t[o] = (_Float16)x.x; t[o+1] = (_Float16)x.y;
    t[o+2] = (_Float16)x.z; t[o+3] = (_Float16)x.w;
}

__global__ __launch_bounds__(256) void k_gemm(GemmParams p) {
    __shared__ __align__(16) _Float16 sA[2 * TILEA];   // 2 x 10 KB
    __shared__ __align__(16) _Float16 sB[2 * TILEB];   // 2 x 10 KB
    const int tid  = threadIdx.x;
    const int wave = tid >> 5;
    const int lane = tid & 31;
    const int g    = blockIdx.z;
    const long m0  = (long)blockIdx.y * TBM;
    const long n0  = (long)blockIdx.x * TBN;
    const float* Ag = p.A  + (long)g * p.a_b;
    const float* Bg = p.Bw + (long)g * p.b_b;

    // staging: thread t covers row (t>>1) of BOTH tiles, k-slice (t&1)*16..+15
    const int  sr   = tid >> 1;            // 0..127
    const int  sc   = (tid & 1) << 4;      // 0 | 16
    const long agm  = m0 + sr;
    const bool amok = agm < (long)p.Mr;
    const long agmc = amok ? agm : (long)(p.Mr - 1);
    const long bgn  = n0 + sr;
    const bool bnok = bgn < (long)p.N;
    const long bgnc = bnok ? bgn : (long)(p.N - 1);
    const float* Arow = Ag + agmc * p.a_m;
    const float* Brow = Bg + bgnc * p.b_n;
    const bool afast = (p.a_k == 1);
    const bool bfast = (p.b_k == 1);

    auto ldA = [&](int k0, v16h& t) {
        if (afast) {
            const float4* v4 = (const float4*)(Arow + k0 + sc);
            cvt4(t, 0, v4[0]); cvt4(t, 4, v4[1]);
            cvt4(t, 8, v4[2]); cvt4(t, 12, v4[3]);
        } else {
            const float* ap = Arow + (long)(k0 + sc) * p.a_k;
            float tmp[16];
            #pragma unroll
            for (int j = 0; j < 16; ++j) tmp[j] = ap[(long)j * p.a_k];
            #pragma unroll
            for (int j = 0; j < 16; ++j) t[j] = (_Float16)tmp[j];
        }
        #pragma unroll
        for (int j = 0; j < 16; ++j) t[j] = amok ? t[j] : (_Float16)0.f;
    };
    auto ldB = [&](int k0, v16h& t) {
        if (bfast) {
            const float4* v4 = (const float4*)(Brow + k0 + sc);
            cvt4(t, 0, v4[0]); cvt4(t, 4, v4[1]);
            cvt4(t, 8, v4[2]); cvt4(t, 12, v4[3]);
        } else {
            const float* bp = Brow + (long)(k0 + sc) * p.b_k;
            float tmp[16];
            #pragma unroll
            for (int j = 0; j < 16; ++j) tmp[j] = bp[(long)j * p.b_k];
            #pragma unroll
            for (int j = 0; j < 16; ++j) t[j] = (_Float16)tmp[j];
        }
        #pragma unroll
        for (int j = 0; j < 16; ++j) t[j] = bnok ? t[j] : (_Float16)0.f;
    };
    auto st = [&](_Float16* base, const v16h& t) {
        v8h lo, hi;
        #pragma unroll
        for (int j = 0; j < 8; ++j) { lo[j] = t[j]; hi[j] = t[j + 8]; }
        *(v8h*)(base + sr * TBKP + sc)     = lo;
        *(v8h*)(base + sr * TBKP + sc + 8) = hi;
    };

    v8f acc[8] = {};

    // prologue: stage first K-slab into buffer 0
    {
        v16h ta, tb;
        ldA(0, ta); ldB(0, tb);
        st(sA, ta); st(sB, tb);
    }
    __syncthreads();

    int cur = 0;
    for (int k0 = 0; k0 < p.Kd; k0 += TBK) {
        const bool more = (k0 + TBK) < p.Kd;
        // issue next slab's global loads before compute (latency hidden by WMMAs)
        v16h na, nb2;
        if (more) {
            ldA(k0 + TBK, na);
            ldB(k0 + TBK, nb2);
            int kn = k0 + 2 * TBK;
            if (kn < p.Kd) {   // global_prefetch_b8 of slab after next
                __builtin_prefetch(Arow + (long)(kn + sc) * p.a_k, 0, 1);
                __builtin_prefetch(Brow + (long)(kn + sc) * p.b_k, 0, 1);
            }
        }

        const _Float16* cA = sA + cur * TILEA;
        const _Float16* cB = sB + cur * TILEB;
        // A fragment: 16x32 f16 (lanes<16: K 0..15; lanes>=16: K 16..31)
        int arow = (wave << 4) + (lane & 15);
        int ksel = (lane >> 4) << 4;
        v16h af;
        #pragma unroll
        for (int i = 0; i < 16; ++i) {
            int kk = i + ksel;
            af[i] = cA[arow * TBKP + kk];
        }
        int kb = (lane >> 4) << 4;   // B frag: lanes<16 K=0..15, lanes>=16 K=16..31
        #pragma unroll
        for (int nb = 0; nb < 8; ++nb) {
            int bcol = (nb << 4) + (lane & 15);
            v16h bf;
            #pragma unroll
            for (int i = 0; i < 16; ++i)
                bf[i] = cB[bcol * TBKP + kb + i];
            acc[nb] = __builtin_amdgcn_wmma_f32_16x16x32_f16(
                false, af, false, bf, (short)0, acc[nb], false, false);
        }

        // write next slab into the other buffer (no conflict with current reads)
        if (more) {
            st(sA + (cur ^ 1) * TILEA, na);
            st(sB + (cur ^ 1) * TILEB, nb2);
        }
        __syncthreads();
        cur ^= 1;
    }

    // ---- epilogue ----
    float* Cg = p.C + (long)g * p.c_b;
    const float* Rg = p.res ? p.res + (long)g * p.r_b : nullptr;
    const bool full = (m0 + TBM <= (long)p.Mr) && (n0 + TBN <= (long)p.N);
    #pragma unroll
    for (int nb = 0; nb < 8; ++nb) {
        #pragma unroll
        for (int r = 0; r < 8; ++r) {
            long mm = m0 + (wave << 4) + r + ((lane >> 4) << 3);
            long nn = n0 + (nb << 4) + (lane & 15);
            if (full || (mm < p.Mr && nn < p.N)) {
                float v = acc[nb][r] * p.scale;
                if (p.bias) v += p.bias[nn];
                if (Rg)     v += Rg[mm * p.r_m + nn * p.r_n];
                if (p.relu) v = v > 0.f ? v : 0.f;
                Cg[mm * p.c_m + nn * p.c_n] = v;
            }
        }
    }
}

// ================= elementwise / small kernels =================
__global__ void k_decomp(const float* __restrict__ x, float* __restrict__ seas,
                         float* __restrict__ tr, int accum, int C, int n) {
    int idx = blockIdx.x * blockDim.x + threadIdx.x;
    if (idx >= n) return;
    int c = idx % C, l = (idx / C) % Lc, b = idx / (C * Lc);
    const int pad = (KMA - 1) / 2;
    long base = (long)b * Lc * C + c;
    float s = 0.f;
    for (int j = -pad; j <= pad; ++j) {
        int ll = l + j; ll = ll < 0 ? 0 : (ll >= Lc ? Lc - 1 : ll);
        s += x[base + (long)ll * C];
    }
    float ma = s / (float)KMA;
    long o0 = base + (long)l * C;
    if (seas) seas[o0] = x[o0] - ma;
    if (tr)   { if (accum) tr[o0] += ma; else tr[o0] = ma; }
}

__global__ void k_embed(const float* __restrict__ x, const float* __restrict__ W,
                        float* __restrict__ out, int n) {
    int idx = blockIdx.x * blockDim.x + threadIdx.x;
    if (idx >= n) return;
    int o = idx % DMc, l = (idx / DMc) % Lc, b = idx / (DMc * Lc);
    const float* xb = x + (long)b * Lc * CIN;
    int lm = (l + Lc - 1) % Lc, lp = (l + 1) % Lc;
    const float* w = W + (long)o * CIN * 3;
    float s = 0.f;
    for (int c = 0; c < CIN; ++c) {
        s += xb[(long)lm * CIN + c] * w[c * 3 + 0]
           + xb[(long)l  * CIN + c] * w[c * 3 + 1]
           + xb[(long)lp * CIN + c] * w[c * 3 + 2];
    }
    int i2 = o & ~1;
    float df  = expf(-(float)i2 * (9.210340371976184f / (float)DMc));
    float ang = (float)l * df;
    s += (o & 1) ? cosf(ang) : sinf(ang);
    out[idx] = s;
}

// DFT bases: bF [2M x L] rows 0..M-1 = cos, M..2M-1 = -sin (rfft)
// bI [2M x L]: irfft truncated to M modes, scaling baked in.
__global__ void k_basis(float* __restrict__ bF, float* __restrict__ bI, int n) {
    int idx = blockIdx.x * blockDim.x + threadIdx.x;
    if (idx >= n) return;
    int t = idx % Lc, m2 = idx / Lc;
    const float w0 = 6.283185307179586f / (float)Lc;
    if (m2 < Mc) {
        float ang = w0 * (float)m2 * (float)t;
        bF[idx] = cosf(ang);
        float sc = (m2 == 0 ? 1.f : 2.f) / (float)Lc;
        bI[idx] = sc * cosf(ang);
    } else {
        int m = m2 - Mc;
        float ang = w0 * (float)m * (float)t;
        bF[idx] = -sinf(ang);
        float sc = (m == 0 ? 0.f : 2.f) / (float)Lc;
        bI[idx] = -sc * sinf(ang);
    }
}

// om[b,h,o,m] = sum_e ft[b,h,e,m] * w[h,e,o,m]   (complex; ft/om in [B,DM,2M] re|im)
__global__ void k_modemm(const float* __restrict__ ft, const float* __restrict__ w,
                         float* __restrict__ om, int n) {
    int idx = blockIdx.x * blockDim.x + threadIdx.x;
    if (idx >= n) return;
    int m = idx % Mc, o = (idx / Mc) % Ec, h = (idx / (Mc * Ec)) % Hc, b = idx / (Mc * Ec * Hc);
    const float* fb = ft + (long)b * DMc * 2 * Mc + (long)h * Ec * 2 * Mc;
    float ar = 0.f, ai = 0.f;
    for (int e = 0; e < Ec; ++e) {
        float xr = fb[(long)e * 2 * Mc + m];
        float xi = fb[(long)e * 2 * Mc + Mc + m];
        long wi = ((((long)h * Ec + e) * Ec + o) * Mc + m) * 2;
        float wr = w[wi], wq = w[wi + 1];
        ar += xr * wr - xi * wq;
        ai += xr * wq + xi * wr;
    }
    long ob = (long)b * DMc * 2 * Mc + ((long)h * Ec + o) * 2 * Mc + m;
    om[ob] = ar; om[ob + Mc] = ai;
}

__device__ inline void cplx_tanh(float a, float b, float& re, float& im) {
    float t = 2.f * a, u = 2.f * b;
    if (fabsf(t) > 40.f) { re = t > 0.f ? 1.f : -1.f; im = 0.f; return; }
    float d = coshf(t) + cosf(u);
    re = sinhf(t) / d; im = sinf(u) / d;
}

// qk[b,h,x,y] = ctanh( sum_e xq[b,h,e,x]*xk[b,h,e,y] )
__global__ void k_qk(const float* __restrict__ xq, const float* __restrict__ xk,
                     float* __restrict__ qk, int n) {
    int idx = blockIdx.x * blockDim.x + threadIdx.x;
    if (idx >= n) return;
    int y = idx % Mc, x = (idx / Mc) % Mc, h = (idx / (Mc * Mc)) % Hc, b = idx / (Mc * Mc * Hc);
    const float* qb = xq + (long)b * DMc * 2 * Mc + (long)h * Ec * 2 * Mc;
    const float* kb = xk + (long)b * DMc * 2 * Mc + (long)h * Ec * 2 * Mc;
    float ar = 0.f, ai = 0.f;
    for (int e = 0; e < Ec; ++e) {
        float qr = qb[(long)e * 2 * Mc + x], qi = qb[(long)e * 2 * Mc + Mc + x];
        float kr = kb[(long)e * 2 * Mc + y], ki = kb[(long)e * 2 * Mc + Mc + y];
        ar += qr * kr - qi * ki;
        ai += qr * ki + qi * kr;
    }
    float tr, ti; cplx_tanh(ar, ai, tr, ti);
    qk[(long)idx * 2] = tr; qk[(long)idx * 2 + 1] = ti;
}

// out[b,h,e,x] = sum_y qk[b,h,x,y]*xk[b,h,e,y]   (-> [B,DM,2M])
__global__ void k_qkv(const float* __restrict__ qk, const float* __restrict__ xk,
                      float* __restrict__ out, int n) {
    int idx = blockIdx.x * blockDim.x + threadIdx.x;
    if (idx >= n) return;
    int x = idx % Mc, e = (idx / Mc) % Ec, h = (idx / (Mc * Ec)) % Hc, b = idx / (Mc * Ec * Hc);
    const float* qb = qk + ((long)b * Hc + h) * Mc * Mc * 2;
    const float* kb = xk + (long)b * DMc * 2 * Mc + ((long)h * Ec + e) * 2 * Mc;
    float ar = 0.f, ai = 0.f;
    for (int y = 0; y < Mc; ++y) {
        float qr = qb[((long)x * Mc + y) * 2], qi = qb[((long)x * Mc + y) * 2 + 1];
        float kr = kb[y], ki = kb[Mc + y];
        ar += qr * kr - qi * ki;
        ai += qr * ki + qi * kr;
    }
    long ob = (long)b * DMc * 2 * Mc + ((long)h * Ec + e) * 2 * Mc + x;
    out[ob] = ar; out[ob + Mc] = ai;
}

// trend[b,l,c] += circ_conv3(tsum, W)  (W: [CIN, DM, 3])
__global__ void k_trendproj(const float* __restrict__ t, const float* __restrict__ W,
                            float* __restrict__ trend, int n) {
    int idx = blockIdx.x * blockDim.x + threadIdx.x;
    if (idx >= n) return;
    int c = idx % CIN, l = (idx / CIN) % Lc, b = idx / (CIN * Lc);
    const float* tb = t + (long)b * Lc * DMc;
    int lm = (l + Lc - 1) % Lc, lp = (l + 1) % Lc;
    const float* w = W + (long)c * DMc * 3;
    float s = 0.f;
    for (int d = 0; d < DMc; ++d) {
        s += tb[(long)lm * DMc + d] * w[d * 3 + 0]
           + tb[(long)l  * DMc + d] * w[d * 3 + 1]
           + tb[(long)lp * DMc + d] * w[d * 3 + 2];
    }
    trend[idx] += s;
}

// ================= host orchestration =================
static inline int cdiv(int a, int b) { return (a + b - 1) / b; }

static void gemm(hipStream_t s, const float* A, const float* Bw, float* C,
                 const float* bias, const float* res,
                 long a_b, long a_m, long a_k,
                 long b_b, long b_n, long b_k,
                 long c_b, long c_m, long c_n,
                 long r_b, long r_m, long r_n,
                 int Mr, int N, int Kd, int G, int relu, float scale) {
    GemmParams p{A, Bw, C, bias, res, a_b, a_m, a_k, b_b, b_n, b_k,
                 c_b, c_m, c_n, r_b, r_m, r_n, Mr, N, Kd, relu, scale};
    dim3 grid(cdiv(N, TBN), cdiv(Mr, TBM), G);
    k_gemm<<<grid, 256, 0, s>>>(p);
}

#define EW(kern, n, ...) kern<<<cdiv((n), 256), 256, 0, stream>>>(__VA_ARGS__)

extern "C" void kernel_launch(void* const* d_in, const int* in_sizes, int n_in,
                              void* d_out, int out_size, void* d_ws, size_t ws_size,
                              hipStream_t stream) {
    (void)in_sizes; (void)n_in; (void)out_size;
    const float* x_enc  = (const float*)d_in[0];
    const float* tok_W  = (const float*)d_in[1];
    const float* eWq = (const float*)d_in[2];
    const float* eWo = (const float*)d_in[5];
    const float* ebq = (const float*)d_in[6];
    const float* ebo = (const float*)d_in[9];
    const float* sWq = (const float*)d_in[10], *sWo = (const float*)d_in[13];
    const float* sbq = (const float*)d_in[14], *sbo = (const float*)d_in[17];
    const float* cWq = (const float*)d_in[18], *cWk = (const float*)d_in[19];
    const float* cWo = (const float*)d_in[21];
    const float* cbq = (const float*)d_in[22], *cbk = (const float*)d_in[23];
    const float* cbo = (const float*)d_in[25];
    const float* ec1W = (const float*)d_in[26], *ec1b = (const float*)d_in[27];
    const float* ec2W = (const float*)d_in[28], *ec2b = (const float*)d_in[29];
    const float* dc1W = (const float*)d_in[30], *dc1b = (const float*)d_in[31];
    const float* dc2W = (const float*)d_in[32], *dc2b = (const float*)d_in[33];
    const float* dtrW = (const float*)d_in[34];
    const float* fb_enc = (const float*)d_in[35];
    const float* fb_dec = (const float*)d_in[36];
    const float* fc_w   = (const float*)d_in[37];
    const float* p1W = (const float*)d_in[38], *p1b = (const float*)d_in[39];
    const float* p2W = (const float*)d_in[40], *p2b = (const float*)d_in[41];
    const float* p3W = (const float*)d_in[42], *p3b = (const float*)d_in[43];
    // K/V projections of the Fourier self-attention and V of the cross-attention
    // are dead code in the reference (fourier_self uses only q; cross uses q,k).

    // workspace carve-up (fp32 elements)
    float* w = (float*)d_ws;
    auto alloc = [&](size_t ne) { float* p = w; w += ne; return p; };
    float* trend = alloc((size_t)BL * CIN);
    float* tsum  = alloc((size_t)BL * DMc);
    float* basF  = alloc((size_t)2 * Mc * Lc);
    float* basI  = alloc((size_t)2 * Mc * Lc);
    float* enc   = alloc((size_t)BL * DMc);
    float* xb    = alloc((size_t)BL * DMc);
    float* b1    = alloc((size_t)BL * DMc);
    float* b2    = alloc((size_t)BL * DMc);
    float* ff    = alloc((size_t)BL * DFFc);
    float* ft1   = alloc((size_t)Bc * DMc * 2 * Mc);
    float* ft2   = alloc((size_t)Bc * DMc * 2 * Mc);
    float* omb   = alloc((size_t)Bc * DMc * 2 * Mc);
    float* qkb   = alloc((size_t)Bc * Hc * Mc * Mc * 2);
    float* o1    = alloc((size_t)BL * CIN);
    float* o2    = alloc((size_t)BL * CIN);
    if ((size_t)((char*)w - (char*)d_ws) > ws_size) return;  // ws too small

    const long FT = (long)DMc * 2 * Mc;   // per-batch spectral block

    // truncated-DFT forward: q [B,L,DM] -> ft [B,DM,2M]
    auto dft_fwd = [&](const float* src, float* dst) {
        gemm(stream, src, basF, dst, nullptr, nullptr,
             (long)Lc * DMc, 1, DMc,          // A[b, d, l]
             0, (long)Lc, 1,                  // B[n(2M), l]
             FT, 2 * Mc, 1,                   // C[b, d, n]
             0, 0, 0, DMc, 2 * Mc, Lc, Bc, 0, 1.f);
    };
    // truncated irFFT: om [B,DM,2M] -> o [B,L,DM]
    auto dft_inv = [&](const float* om, float* dst, float scale) {
        gemm(stream, om, basI, dst, nullptr, nullptr,
             FT, 2 * Mc, 1,                   // A[b, d, k(2M)]
             0, 1, (long)Lc,                  // B[n=l, k] = basI[k*Lc + l]
             (long)Lc * DMc, 1, DMc,          // C[b, l, d] with rows=d
             0, 0, 0, DMc, Lc, 2 * Mc, Bc, 0, scale);
    };
    // dense projection: C[BL,N] = A[BL,K] @ W[N,K]^T + bias (+res) (relu?)
    auto proj = [&](const float* A, const float* W_, const float* bias,
                    const float* res, float* C, int N, int Kd, int relu) {
        gemm(stream, A, W_, C, bias, res,
             0, (long)Kd, 1, 0, (long)Kd, 1, 0, (long)N, 1, 0, (long)N, 1,
             BL, N, Kd, 1, relu, 1.f);
    };

    // ---- prologue ----
    EW(k_decomp, BL * CIN, x_enc, nullptr, trend, 0, CIN, BL * CIN);  // trend_init
    EW(k_basis, 2 * Mc * Lc, basF, basI, 2 * Mc * Lc);
    EW(k_embed, BL * DMc, x_enc, tok_W, enc, BL * DMc);

    // ---- encoder ----
    for (int l = 0; l < ELc; ++l) {
        const float* Wq = eWq + (long)l * DMc * DMc;
        const float* Wo = eWo + (long)l * DMc * DMc;
        proj(enc, Wq, ebq + (long)l * DMc, nullptr, b1, DMc, DMc, 0);
        dft_fwd(b1, ft1);
        EW(k_modemm, Bc * DMc * Mc, ft1, fb_enc + (long)l * Hc * Ec * Ec * Mc * 2, omb, Bc * DMc * Mc);
        dft_inv(omb, b2, 1.f);
        proj(b2, Wo, ebo + (long)l * DMc, enc, b1, DMc, DMc, 0);     // h = enc + mha
        EW(k_decomp, BL * DMc, b1, b2, nullptr, 0, DMc, BL * DMc);   // h = seasonal(h)
        proj(b2, ec1W + (long)l * DFFc * DMc, ec1b + (long)l * DFFc, nullptr, ff, DFFc, DMc, 1);
        proj(ff, ec2W + (long)l * DMc * DFFc, ec2b + (long)l * DMc, b2, b1, DMc, DFFc, 0); // h+y
        EW(k_decomp, BL * DMc, b1, enc, nullptr, 0, DMc, BL * DMc);  // enc = seasonal(h+y)
    }

    // ---- decoder (x starts as enc_out; dec embedding is dead code upstream) ----
    hipMemcpyAsync(xb, enc, (size_t)BL * DMc * sizeof(float), hipMemcpyDeviceToDevice, stream);
    hipMemsetAsync(tsum, 0, (size_t)BL * DMc * sizeof(float), stream);

    // self-attention (Fourier)
    proj(xb, sWq, sbq, nullptr, b1, DMc, DMc, 0);
    dft_fwd(b1, ft1);
    EW(k_modemm, Bc * DMc * Mc, ft1, fb_dec, omb, Bc * DMc * Mc);
    dft_inv(omb, b2, 1.f);
    proj(b2, sWo, sbo, xb, b1, DMc, DMc, 0);                          // x + mha
    EW(k_decomp, BL * DMc, b1, xb, tsum, 1, DMc, BL * DMc);           // x, t1 -> tsum

    // cross-attention (Fourier, complex tanh; k-modes used as values)
    proj(xb, cWq, cbq, nullptr, b1, DMc, DMc, 0);
    proj(enc, cWk, cbk, nullptr, b2, DMc, DMc, 0);
    dft_fwd(b1, ft1);
    dft_fwd(b2, ft2);
    EW(k_qk,  Bc * Hc * Mc * Mc, ft1, ft2, qkb, Bc * Hc * Mc * Mc);
    EW(k_qkv, Bc * Hc * Ec * Mc, qkb, ft2, ft1, Bc * Hc * Ec * Mc);
    EW(k_modemm, Bc * DMc * Mc, ft1, fc_w, omb, Bc * DMc * Mc);
    dft_inv(omb, b2, 1.f / ((float)DMc * (float)DMc));
    proj(b2, cWo, cbo, xb, b1, DMc, DMc, 0);                          // x + cross
    EW(k_decomp, BL * DMc, b1, xb, tsum, 1, DMc, BL * DMc);           // x, t2

    // FFN
    proj(xb, dc1W, dc1b, nullptr, ff, DFFc, DMc, 1);
    proj(ff, dc2W, dc2b, xb, b1, DMc, DFFc, 0);                       // x + y
    EW(k_decomp, BL * DMc, b1, xb, tsum, 1, DMc, BL * DMc);           // x, t3

    // trend += circ_conv3(t1+t2+t3, dec_trW)
    EW(k_trendproj, BL * CIN, tsum, dtrW, trend, BL * CIN);

    // ---- heads ----
    proj(xb, p3W, p3b, trend, o1, CIN, DMc, 0);        // out1 = trend + x@p3W^T + p3b
    proj(o1, p1W, p1b, nullptr, o2, CIN, CIN, 0);      // out2 = out1@p1W^T + p1b
    // out3[b,p,c] = sum_l out2[b,l,c]*p2W[p,l] + p2b[p]   (rows=c, n=p, k=l)
    gemm(stream, o2, p2W, (float*)d_out, p2b, nullptr,
         (long)Lc * CIN, 1, (long)CIN,          // A[b, c, l]
         0, (long)Lc, 1,                        // B[p, l]
         (long)PRED * CIN, 1, (long)CIN,        // C[b, p, c] rows=c
         0, 0, 0, CIN, PRED, Lc, Bc, 0, 1.f);
}
